// CrossTransformer_62560493633844
// MI455X (gfx1250) — matrix operand order repaired and checked by reading, not compiled, not run on hardware
//
#include <hip/hip_runtime.h>

// ---------------------------------------------------------------------------
// CrossTransformer block on MI455X (gfx1250, wave32, WMMA bf16->f32).
//   1) convert x -> bf16; convert+TRANSPOSE weights -> bf16 [N][K]
//   2) proj GEMMs: qk0,qk1 (scaled DH^-0.25), v0,v1
//   3) bisoftmax as two flash attentions (dir axis), no sim materialization
//   4) out-proj GEMMs -> mo0, mo1
//   5) FFN1 (concat GEMM K=512, full-row blocks) + LayerNorm + exact GELU
//   6) FFN2 GEMM + bias + fp32 residual -> d_out
// Fragments: every WMMA operand is two ds_load_b128s. Tile staging uses
// CDNA5 async global->LDS (ASYNCcnt) with LDS double buffering in the GEMM.
// ---------------------------------------------------------------------------

typedef __attribute__((ext_vector_type(16))) __bf16 v16bf;
typedef __attribute__((ext_vector_type(8)))  __bf16 v8bf;
typedef __attribute__((ext_vector_type(8)))  float  v8f;
typedef int v4i __attribute__((vector_size(16)));

#define DEV static __device__ __forceinline__
#define AS1 __attribute__((address_space(1)))
#define AS3 __attribute__((address_space(3)))

#ifndef __has_builtin
#define __has_builtin(x) 0
#endif
#if __has_builtin(__builtin_amdgcn_global_load_async_to_lds_b128)
#define ASYNC_LDS 1
#else
#define ASYNC_LDS 0
#endif

static constexpr int Bb = 4, Nn = 2048, Dd = 256, Hh = 4, Dh = 64;
static constexpr int Mrows = Bb * Nn;          // 8192
static constexpr float QK_SCALE = 0.35355339059327373f;  // DH^-0.25

DEV v8f wmma_bf16(v16bf a, v16bf b, v8f c) {
  return __builtin_amdgcn_wmma_f32_16x16x32_bf16(false, a, false, b,
                                                 (short)0, c, false, false);
}
DEV v8f v8f_zero() { v8f z = {}; return z; }

// 16-byte global->LDS copy: async on CDNA5 (ASYNCcnt), sync fallback.
DEV void async_cp_b128(const __bf16* g, __bf16* l) {
#if ASYNC_LDS
  __builtin_amdgcn_global_load_async_to_lds_b128((AS1 v4i*)g, (AS3 v4i*)l,
                                                 0, 0);
#else
  *(uint4*)l = *(const uint4*)g;
#endif
}
DEV void wait_async0() {
#if ASYNC_LDS
#if __has_builtin(__builtin_amdgcn_s_wait_asynccnt)
  __builtin_amdgcn_s_wait_asynccnt(0);
#else
  asm volatile("s_wait_asynccnt 0" ::: "memory");
#endif
#endif
}

DEV v16bf frag_cat(v8bf lo, v8bf hi) {
  return __builtin_shufflevector(lo, hi, 0, 1, 2, 3, 4, 5, 6, 7,
                                 8, 9, 10, 11, 12, 13, 14, 15);
}

// A fragment: 16x32 bf16 (MxK), LDS tile row-major [m][k], ld*2 % 16 == 0.
DEV v16bf frag_a(const __bf16* p, int ld, int lane) {
  int row = lane & 15;
  int kb  = (lane >> 4) << 3;
  const __bf16* q = p + row * ld + kb;
  v8bf lo = *(const v8bf*)(q);        // K kb..kb+7     (ds_load_b128)
  v8bf hi = *(const v8bf*)(q + 16);   // K kb+16..kb+23 (ds_load_b128)
  return frag_cat(lo, hi);
}

// B fragment: logical B[k][n] (32x16) with the tile stored [n][k]-major.
DEV v16bf frag_bnk(const __bf16* p, int ld, int lane) {
  int col = lane & 15;
  int kb  = (lane >> 4) << 3;
  const __bf16* q = p + col * ld + kb;
  v8bf lo = *(const v8bf*)(q);
  v8bf hi = *(const v8bf*)(q + 16);
  return frag_cat(lo, hi);
}

// ---------------------------------------------------------------------------
__global__ void f32_to_bf16_kernel(const float* __restrict__ in,
                                   __bf16* __restrict__ out, int n) {
  int i = blockIdx.x * blockDim.x + threadIdx.x;
  int stride = gridDim.x * blockDim.x;
  for (; i < n; i += stride) out[i] = (__bf16)in[i];
}

// in: [K][N] row-major f32 -> out: [N][K] row-major bf16 (weight transpose)
__global__ void f32_to_bf16_T_kernel(const float* __restrict__ in,
                                     __bf16* __restrict__ out, int K, int N) {
  int i = blockIdx.x * blockDim.x + threadIdx.x;
  int stride = gridDim.x * blockDim.x;
  int total = K * N;
  for (; i < total; i += stride) {
    int k = i / N, n = i - k * N;
    out[(size_t)n * K + k] = (__bf16)in[i];
  }
}

// ---------------------------------------------------------------------------
// Generic bf16 GEMM, weights pre-transposed BwT[N][K].
// Block: 128x64 tile, 8 waves; wave = 16 rows x 64 cols; K step 32;
// double-buffered LDS fed by async global->LDS copies.
constexpr int LA = 40, LBT = 40;
constexpr int ATILE = 128 * LA, BTILE = 64 * LBT;

DEV void gemm_stage(const __bf16* A, int lda, const __bf16* BwT, int K,
                    int row0, int col0, int kt, __bf16* At, __bf16* Bt,
                    int tid) {
#pragma unroll
  for (int g = 0; g < 2; ++g) {          // A tile 128x32: 512 b128 chunks
    int idx = (tid * 2 + g) * 8;
    int r = idx >> 5, c = idx & 31;
    async_cp_b128(A + (size_t)(row0 + r) * lda + kt + c, At + r * LA + c);
  }
  {                                      // B tile [64 n][32 k]: 256 chunks
    int idx = tid * 8;
    int n = idx >> 5, c = idx & 31;
    async_cp_b128(BwT + (size_t)(col0 + n) * K + kt + c, Bt + n * LBT + c);
  }
}

__global__ __launch_bounds__(256)
void gemm_bf16_kernel(const __bf16* __restrict__ A, int lda,
                      const __bf16* __restrict__ BwT,   // [N][K]
                      const float* __restrict__ bias,
                      __bf16* __restrict__ outb,
                      float* __restrict__ outf,
                      const float* __restrict__ residual,
                      int M, int N, int K, float scale) {
  __shared__ __align__(16) __bf16 At[2 * ATILE];   // [m][k] x2
  __shared__ __align__(16) __bf16 Bt[2 * BTILE];   // [n][k] x2

  const int tid = threadIdx.x, lane = tid & 31, wave = tid >> 5;
  const int row0 = blockIdx.x * 128;
  const int col0 = blockIdx.y * 64;

  v8f acc[4];
#pragma unroll
  for (int t = 0; t < 4; ++t) acc[t] = v8f_zero();

  gemm_stage(A, lda, BwT, K, row0, col0, 0, At, Bt, tid);
  wait_async0();
  __syncthreads();

  int buf = 0;
  for (int kt = 0; kt < K; kt += 32) {
    const bool has_next = (kt + 32 < K);
    if (has_next)
      gemm_stage(A, lda, BwT, K, row0, col0, kt + 32, At + (buf ^ 1) * ATILE,
                 Bt + (buf ^ 1) * BTILE, tid);

    v16bf a = frag_a(&At[buf * ATILE + (wave * 16) * LA], LA, lane);
#pragma unroll
    for (int t = 0; t < 4; ++t) {
      v16bf b = frag_bnk(&Bt[buf * BTILE + (t * 16) * LBT], LBT, lane);
      acc[t] = wmma_bf16(a, b, acc[t]);
    }

    if (has_next) {
      wait_async0();
      __syncthreads();
      buf ^= 1;
    }
  }

  const int rb = (lane >> 4) << 3;
  const int cl = lane & 15;
#pragma unroll
  for (int t = 0; t < 4; ++t) {
#pragma unroll
    for (int v = 0; v < 8; ++v) {
      int r = row0 + wave * 16 + rb + v;
      int c = col0 + t * 16 + cl;
      float val = (acc[t][v] + bias[c]) * scale;
      if (residual) val += residual[(size_t)r * N + c];
      if (outb) outb[(size_t)r * N + c] = (__bf16)val;
      else      outf[(size_t)r * N + c] = val;
    }
  }
}

// ---------------------------------------------------------------------------
// Bisoftmax flash attention. dir=0: m0 = softmax_row(qk0 qk1^T) v1
//                            dir=1: m1 = softmax_row(qk1 qk0^T) v0
// Q/K tiles staged with async global->LDS; V transposed to [dh][j] at store.
__global__ __launch_bounds__(256)
void attn_bisoftmax_kernel(const __bf16* __restrict__ qk0,
                           const __bf16* __restrict__ qk1,
                           const __bf16* __restrict__ v0,
                           const __bf16* __restrict__ v1,
                           __bf16* __restrict__ m0,
                           __bf16* __restrict__ m1) {
  constexpr int LQ = 72, LK = 72, LV = 72, LP = 72;
  __shared__ __align__(16) __bf16 Qt[128 * LQ];      // [m][d]
  __shared__ __align__(16) __bf16 Kt[64 * LK];       // [j][d]
  __shared__ __align__(16) __bf16 VtT[64 * LV];      // [dh][j]
  __shared__ __align__(16) __bf16 Pt[8 * 16 * LP];   // per-wave [m][j]

  const int tid = threadIdx.x, lane = tid & 31, wave = tid >> 5;
  const int q0  = blockIdx.x * 128;
  const int b   = blockIdx.y / Hh, h = blockIdx.y % Hh;
  const int dir = blockIdx.z;

  const __bf16* Q  = dir ? qk1 : qk0;
  const __bf16* Kp = dir ? qk0 : qk1;
  const __bf16* Vp = dir ? v0  : v1;
  __bf16*       Om = dir ? m1  : m0;

  const size_t baseRow = (size_t)b * Nn;
  const int    cbase   = h * Dh;

  // Q tile 128x64 async: 1024 b128 chunks, 4 / thread
#pragma unroll
  for (int g = 0; g < 4; ++g) {
    int idx = (tid * 4 + g) * 8;
    int r = idx >> 6, c = idx & 63;
    async_cp_b128(Q + (baseRow + q0 + r) * Dd + cbase + c, &Qt[r * LQ + c]);
  }

  v8f o[4];
#pragma unroll
  for (int t = 0; t < 4; ++t) o[t] = v8f_zero();
  float mrow[8], lrow[8];
#pragma unroll
  for (int v = 0; v < 8; ++v) { mrow[v] = -1e30f; lrow[v] = 0.f; }

  const int rb = (lane >> 4) << 3;
  const int cl = lane & 15;
  __bf16* Pw = &Pt[wave * 16 * LP];

  for (int j0 = 0; j0 < Nn; j0 += 64) {
    __syncthreads();
    // K tile async: 512 b128 chunks, 2 / thread
#pragma unroll
    for (int g = 0; g < 2; ++g) {
      int idx = (tid * 2 + g) * 8;
      int r = idx >> 6, c = idx & 63;
      async_cp_b128(Kp + (baseRow + j0 + r) * Dd + cbase + c,
                    &Kt[r * LK + c]);
    }
    // V tile: transpose into [dh][j] (synchronous, element stores)
#pragma unroll
    for (int g = 0; g < 4; ++g) {
      int idx = (tid << 4) + (g << 2);
      int r = idx >> 6, c = idx & 63;      // r = j row, c = dh col
      uint2 vv = *(const uint2*)(Vp + (baseRow + j0 + r) * Dd + cbase + c);
      const __bf16* ve = (const __bf16*)&vv;
#pragma unroll
      for (int e = 0; e < 4; ++e) VtT[(c + e) * LV + r] = ve[e];
    }
    wait_async0();   // Q (first iter) + K landed
    __syncthreads();

    // S = Q K^T (16 x 64 per wave)
    v8f s[4];
#pragma unroll
    for (int t = 0; t < 4; ++t) s[t] = v8f_zero();
#pragma unroll
    for (int kk = 0; kk < 64; kk += 32) {
      v16bf a = frag_a(&Qt[(wave * 16) * LQ + kk], LQ, lane);
#pragma unroll
      for (int t = 0; t < 4; ++t) {
        v16bf bb = frag_bnk(&Kt[(t * 16) * LK + kk], LK, lane);
        s[t] = wmma_bf16(a, bb, s[t]);
      }
    }

    // Online softmax: row stats reduced across each 16-lane half.
#pragma unroll
    for (int v = 0; v < 8; ++v) {
      float tm = fmaxf(fmaxf(s[0][v], s[1][v]), fmaxf(s[2][v], s[3][v]));
      tm = fmaxf(tm, __shfl_xor(tm, 1, 32));
      tm = fmaxf(tm, __shfl_xor(tm, 2, 32));
      tm = fmaxf(tm, __shfl_xor(tm, 4, 32));
      tm = fmaxf(tm, __shfl_xor(tm, 8, 32));
      float mn   = fmaxf(mrow[v], tm);
      float corr = __expf(mrow[v] - mn);
      float rs = 0.f;
#pragma unroll
      for (int t = 0; t < 4; ++t) {
        float p = __expf(s[t][v] - mn);
        s[t][v] = p;
        rs += p;
      }
      rs += __shfl_xor(rs, 1, 32);
      rs += __shfl_xor(rs, 2, 32);
      rs += __shfl_xor(rs, 4, 32);
      rs += __shfl_xor(rs, 8, 32);
      lrow[v] = lrow[v] * corr + rs;
      mrow[v] = mn;
#pragma unroll
      for (int t = 0; t < 4; ++t) o[t][v] *= corr;
    }

    // Stage P (C-layout -> A-layout) through per-wave LDS.
#pragma unroll
    for (int t = 0; t < 4; ++t)
#pragma unroll
      for (int v = 0; v < 8; ++v)
        Pw[(rb + v) * LP + t * 16 + cl] = (__bf16)s[t][v];
    asm volatile("s_wait_dscnt 0" ::: "memory");  // intra-wave LDS RAW

    // O += P V
#pragma unroll
    for (int kk = 0; kk < 64; kk += 32) {
      v16bf a = frag_a(&Pw[kk], LP, lane);
#pragma unroll
      for (int t = 0; t < 4; ++t) {
        v16bf bb = frag_bnk(&VtT[(t * 16) * LV + kk], LV, lane);
        o[t] = wmma_bf16(a, bb, o[t]);
      }
    }
  }

  // Normalize and store merged-head output.
#pragma unroll
  for (int t = 0; t < 4; ++t)
#pragma unroll
    for (int v = 0; v < 8; ++v) {
      float val = o[t][v] / lrow[v];
      Om[(baseRow + q0 + wave * 16 + rb + v) * Dd + cbase + t * 16 + cl] =
          (__bf16)val;
    }
}

// ---------------------------------------------------------------------------
// FFN1: h = gelu(layernorm(concat(x, mo) @ w_f1 + b_f1)).
// Block = 16 rows x all 512 cols (8 waves x 64-col strips) so LN is local.
// Tiles staged via async global->LDS (single-buffered; B tile is 40 KB).
__global__ __launch_bounds__(256)
void ffn1_ln_gelu_kernel(const __bf16* __restrict__ x0b,
                         const __bf16* __restrict__ mo0,
                         const __bf16* __restrict__ x1b,
                         const __bf16* __restrict__ mo1,
                         const __bf16* __restrict__ wf1T,  // [512][512]
                         const float* __restrict__ bf1,
                         const float* __restrict__ lng,
                         const float* __restrict__ lnb,
                         __bf16* __restrict__ h0,
                         __bf16* __restrict__ h1) {
  constexpr int K = 512, Nf = 512;
  constexpr int LH = 520;
  __shared__ __align__(16) __bf16 At[16 * LA];
  __shared__ __align__(16) unsigned char smem_u[512 * LBT * 2];  // 40960 B
  __bf16* BtT = (__bf16*)smem_u;            // [512 n][40] during GEMM
  float*  Hst = (float*)smem_u;             // [16][520] f32 during LN (reuse)

  const int tid = threadIdx.x, lane = tid & 31, wave = tid >> 5;
  const int row0 = blockIdx.x * 16;
  const int strm = blockIdx.z;
  const __bf16* xb = strm ? x1b : x0b;
  const __bf16* mo = strm ? mo1 : mo0;
  __bf16*       ho = strm ? h1  : h0;

  v8f acc[4];
#pragma unroll
  for (int t = 0; t < 4; ++t) acc[t] = v8f_zero();

  for (int kt = 0; kt < K; kt += 32) {
    __syncthreads();
    // A tile 16x32 from concat(x, mo): 64 b128 chunks (tid < 64)
    if (tid < 64) {
      int idx = tid * 8;
      int r = idx >> 5, c = idx & 31;
      int kc = kt + c;                       // chunk never straddles Dd
      const __bf16* src = (kc < Dd)
          ? (xb + (size_t)(row0 + r) * Dd + kc)
          : (mo + (size_t)(row0 + r) * Dd + (kc - Dd));
      async_cp_b128(src, &At[r * LA + c]);
    }
    // B tile [512 n][32 k]: 2048 b128 chunks, 8 / thread
#pragma unroll
    for (int g = 0; g < 8; ++g) {
      int idx = (tid * 8 + g) * 8;
      int n = idx >> 5, c = idx & 31;
      async_cp_b128(wf1T + (size_t)n * K + kt + c, &BtT[n * LBT + c]);
    }
    wait_async0();
    __syncthreads();

    v16bf a = frag_a(&At[0], LA, lane);
#pragma unroll
    for (int t = 0; t < 4; ++t) {
      v16bf b = frag_bnk(&BtT[(wave * 64 + t * 16) * LBT], LBT, lane);
      acc[t] = wmma_bf16(a, b, acc[t]);
    }
  }

  __syncthreads();  // all BtT reads done before reusing smem as Hst
  const int rb = (lane >> 4) << 3;
  const int cl = lane & 15;
#pragma unroll
  for (int t = 0; t < 4; ++t)
#pragma unroll
    for (int v = 0; v < 8; ++v) {
      int c = wave * 64 + t * 16 + cl;
      Hst[(rb + v) * LH + c] = acc[t][v] + bf1[c];
    }
  __syncthreads();

  // LayerNorm + exact GELU: each wave owns 2 rows of 512.
#pragma unroll
  for (int rr = 0; rr < 2; ++rr) {
    int r = wave * 2 + rr;
    float sum = 0.f;
#pragma unroll
    for (int i = 0; i < 16; ++i) sum += Hst[r * LH + lane + i * 32];
    sum += __shfl_xor(sum, 1, 32);  sum += __shfl_xor(sum, 2, 32);
    sum += __shfl_xor(sum, 4, 32);  sum += __shfl_xor(sum, 8, 32);
    sum += __shfl_xor(sum, 16, 32);
    float mu = sum * (1.f / 512.f);
    float vs = 0.f;
#pragma unroll
    for (int i = 0; i < 16; ++i) {
      float d = Hst[r * LH + lane + i * 32] - mu;
      vs += d * d;
    }
    vs += __shfl_xor(vs, 1, 32);  vs += __shfl_xor(vs, 2, 32);
    vs += __shfl_xor(vs, 4, 32);  vs += __shfl_xor(vs, 8, 32);
    vs += __shfl_xor(vs, 16, 32);
    float rstd = rsqrtf(vs * (1.f / 512.f) + 1e-5f);
#pragma unroll
    for (int i = 0; i < 16; ++i) {
      int c = lane + i * 32;
      float val = (Hst[r * LH + c] - mu) * rstd * lng[c] + lnb[c];
      val = 0.5f * val * (1.f + erff(val * 0.70710678118654752f));
      ho[(size_t)(row0 + r) * Nf + c] = (__bf16)val;
    }
  }
}

// ---------------------------------------------------------------------------
extern "C" void kernel_launch(void* const* d_in, const int* in_sizes, int n_in,
                              void* d_out, int out_size, void* d_ws,
                              size_t ws_size, hipStream_t stream) {
  const float* x0   = (const float*)d_in[0];
  const float* x1   = (const float*)d_in[1];
  const float* w_qk = (const float*)d_in[2];
  const float* b_qk = (const float*)d_in[3];
  const float* w_v  = (const float*)d_in[4];
  const float* b_v  = (const float*)d_in[5];
  const float* w_out= (const float*)d_in[6];
  const float* b_out= (const float*)d_in[7];
  const float* w_f1 = (const float*)d_in[8];
  const float* b_f1 = (const float*)d_in[9];
  const float* ln_g = (const float*)d_in[10];
  const float* ln_b = (const float*)d_in[11];
  const float* w_f2 = (const float*)d_in[12];
  const float* b_f2 = (const float*)d_in[13];

  char* ws = (char*)d_ws;
  size_t off = 0;
  auto take = [&](size_t bytes) {
    void* p = ws + off;
    off += (bytes + 255) & ~(size_t)255;
    return p;
  };
  const size_t MD  = (size_t)Mrows * Dd;   // 8192*256
  const size_t M2D = (size_t)Mrows * 2 * Dd;

  __bf16* xb0   = (__bf16*)take(MD * 2);
  __bf16* xb1   = (__bf16*)take(MD * 2);
  __bf16* wqkT  = (__bf16*)take((size_t)Dd * Dd * 2);
  __bf16* wvT   = (__bf16*)take((size_t)Dd * Dd * 2);
  __bf16* woutT = (__bf16*)take((size_t)Dd * Dd * 2);
  __bf16* wf1T  = (__bf16*)take((size_t)2 * Dd * 2 * Dd * 2);
  __bf16* wf2T  = (__bf16*)take((size_t)2 * Dd * Dd * 2);
  __bf16* qk0   = (__bf16*)take(MD * 2);
  __bf16* qk1   = (__bf16*)take(MD * 2);
  __bf16* v0    = (__bf16*)take(MD * 2);
  __bf16* v1    = (__bf16*)take(MD * 2);
  __bf16* m0    = (__bf16*)take(MD * 2);
  __bf16* m1    = (__bf16*)take(MD * 2);
  __bf16* mo0   = (__bf16*)take(MD * 2);
  __bf16* mo1   = (__bf16*)take(MD * 2);
  __bf16* h0    = (__bf16*)take(M2D * 2);
  __bf16* h1    = (__bf16*)take(M2D * 2);

  float* out0 = (float*)d_out;
  float* out1 = out0 + MD;

  // 1) conversions (x straight, weights transposed to [N][K])
  f32_to_bf16_kernel<<<dim3(256), dim3(256), 0, stream>>>(x0, xb0, (int)MD);
  f32_to_bf16_kernel<<<dim3(256), dim3(256), 0, stream>>>(x1, xb1, (int)MD);
  f32_to_bf16_T_kernel<<<dim3(128), dim3(256), 0, stream>>>(w_qk, wqkT, Dd, Dd);
  f32_to_bf16_T_kernel<<<dim3(128), dim3(256), 0, stream>>>(w_v, wvT, Dd, Dd);
  f32_to_bf16_T_kernel<<<dim3(128), dim3(256), 0, stream>>>(w_out, woutT, Dd, Dd);
  f32_to_bf16_T_kernel<<<dim3(256), dim3(256), 0, stream>>>(w_f1, wf1T,
                                                            2 * Dd, 2 * Dd);
  f32_to_bf16_T_kernel<<<dim3(128), dim3(256), 0, stream>>>(w_f2, wf2T,
                                                            2 * Dd, Dd);

  // 2) projections (scaled qk, plain v)
  dim3 ggrid(Mrows / 128, Dd / 64), gblk(256);
  gemm_bf16_kernel<<<ggrid, gblk, 0, stream>>>(xb0, Dd, wqkT, b_qk, qk0,
                                               nullptr, nullptr, Mrows, Dd, Dd,
                                               QK_SCALE);
  gemm_bf16_kernel<<<ggrid, gblk, 0, stream>>>(xb1, Dd, wqkT, b_qk, qk1,
                                               nullptr, nullptr, Mrows, Dd, Dd,
                                               QK_SCALE);
  gemm_bf16_kernel<<<ggrid, gblk, 0, stream>>>(xb0, Dd, wvT, b_v, v0,
                                               nullptr, nullptr, Mrows, Dd, Dd,
                                               1.0f);
  gemm_bf16_kernel<<<ggrid, gblk, 0, stream>>>(xb1, Dd, wvT, b_v, v1,
                                               nullptr, nullptr, Mrows, Dd, Dd,
                                               1.0f);

  // 3) bisoftmax flash attention, both directions
  attn_bisoftmax_kernel<<<dim3(Nn / 128, Bb * Hh, 2), gblk, 0, stream>>>(
      qk0, qk1, v0, v1, m0, m1);

  // 4) output projections
  gemm_bf16_kernel<<<ggrid, gblk, 0, stream>>>(m0, Dd, woutT, b_out, mo0,
                                               nullptr, nullptr, Mrows, Dd, Dd,
                                               1.0f);
  gemm_bf16_kernel<<<ggrid, gblk, 0, stream>>>(m1, Dd, woutT, b_out, mo1,
                                               nullptr, nullptr, Mrows, Dd, Dd,
                                               1.0f);

  // 5) FFN1 + LN + GELU (z axis = stream)
  ffn1_ln_gelu_kernel<<<dim3(Mrows / 16, 1, 2), gblk, 0, stream>>>(
      xb0, mo0, xb1, mo1, wf1T, b_f1, ln_g, ln_b, h0, h1);

  // 6) FFN2 + bias + fp32 residual -> outputs
  gemm_bf16_kernel<<<ggrid, gblk, 0, stream>>>(h0, 2 * Dd, wf2T, b_f2,
                                               nullptr, out0, x0, Mrows, Dd,
                                               2 * Dd, 1.0f);
  gemm_bf16_kernel<<<ggrid, gblk, 0, stream>>>(h1, 2 * Dd, wf2T, b_f2,
                                               nullptr, out1, x1, Mrows, Dd,
                                               2 * Dd, 1.0f);
}
